// PointMLPModel_3118146257064
// MI455X (gfx1250) — matrix operand order repaired and checked by reading, not compiled, not run on hardware
//
#include <hip/hip_runtime.h>
#include <math.h>

#define EPSV 1e-5f

typedef __attribute__((ext_vector_type(16))) __bf16 v16bf;
typedef __attribute__((ext_vector_type(8)))  float  v8f;
typedef __attribute__((ext_vector_type(8)))  unsigned short us8;
typedef __attribute__((ext_vector_type(16))) unsigned short us16;
typedef unsigned short u16;

// ---------- bf16 helpers (bit-level, RNE) ----------
__device__ __forceinline__ u16 f2bf_bits(float f) {
  unsigned u = __float_as_uint(f);
  unsigned r = u + 0x7FFFu + ((u >> 16) & 1u);
  return (u16)(r >> 16);
}
__device__ __forceinline__ float bf2f(u16 s) {
  return __uint_as_float(((unsigned)s) << 16);
}

// ---------- f32 -> bf16 weight conversion ----------
__global__ void convert_f32_bf16_kernel(const float* __restrict__ src,
                                        u16* __restrict__ dst, long n) {
  long i = (long)blockIdx.x * blockDim.x + threadIdx.x;
  if (i < n) dst[i] = f2bf_bits(src[i]);
}

// ---------- embed: conv(3->64)+BN+ReLU, also writes xyz (point-major) ----------
__global__ void embed_kernel(const float* __restrict__ x, const float* __restrict__ W,
                             const float* __restrict__ bb, const float* __restrict__ bg,
                             const float* __restrict__ bm, const float* __restrict__ bv,
                             u16* __restrict__ h, float* __restrict__ xyz) {
  int t = blockIdx.x * blockDim.x + threadIdx.x; // b*1024 + p
  if (t >= 16 * 1024) return;
  int b = t >> 10, p = t & 1023;
  float x0 = x[b * 3072 + p];
  float x1 = x[b * 3072 + 1024 + p];
  float x2 = x[b * 3072 + 2048 + p];
  xyz[t * 3 + 0] = x0; xyz[t * 3 + 1] = x1; xyz[t * 3 + 2] = x2;
  for (int o = 0; o < 64; ++o) {
    float a = W[o * 3] * x0 + W[o * 3 + 1] * x1 + W[o * 3 + 2] * x2;
    float s = bg[o] * rsqrtf(bv[o] + EPSV);
    float v = a * s + (bb[o] - bm[o] * s);
    h[(long)t * 64 + o] = f2bf_bits(fmaxf(v, 0.f));
  }
}

// ---------- farthest point sampling: one block per batch ----------
__global__ void fps_kernel(const float* __restrict__ xyz, int n, int npoint,
                           int* __restrict__ cent) {
  int b = blockIdx.x, tid = threadIdx.x;
  const int T = 256;
  __shared__ float dist[1024];
  __shared__ float rv[256];
  __shared__ int   ri[256];
  for (int p = tid; p < n; p += T) dist[p] = 1e10f;
  __syncthreads();
  int far = 0;
  for (int i = 0; i < npoint; ++i) {
    if (tid == 0) cent[b * npoint + i] = far;
    float cx = xyz[(b * n + far) * 3 + 0];
    float cy = xyz[(b * n + far) * 3 + 1];
    float cz = xyz[(b * n + far) * 3 + 2];
    float bv2 = -1.f; int bp = 0;
    for (int p = tid; p < n; p += T) {
      float dx = xyz[(b * n + p) * 3 + 0] - cx;
      float dy = xyz[(b * n + p) * 3 + 1] - cy;
      float dz = xyz[(b * n + p) * 3 + 2] - cz;
      float d = dx * dx + dy * dy + dz * dz;
      float nd = fminf(dist[p], d);
      dist[p] = nd;
      if (nd > bv2) { bv2 = nd; bp = p; }
    }
    rv[tid] = bv2; ri[tid] = bp;
    __syncthreads();
    for (int s = 128; s > 0; s >>= 1) {
      if (tid < s) {
        if (rv[tid + s] > rv[tid] ||
            (rv[tid + s] == rv[tid] && ri[tid + s] < ri[tid])) {
          rv[tid] = rv[tid + s]; ri[tid] = ri[tid + s];
        }
      }
      __syncthreads();
    }
    far = ri[0];
    __syncthreads();
  }
}

// ---------- KNN (K=24) + gather new_xyz ----------
__global__ void knn_kernel(const float* __restrict__ xyz, const int* __restrict__ fi,
                           int n, int nnew, float* __restrict__ nxyz,
                           int* __restrict__ idx) {
  int t = blockIdx.x * blockDim.x + threadIdx.x;
  if (t >= 16 * nnew) return;
  int b = t / nnew;
  int f = fi[t];
  float qx = xyz[(b * n + f) * 3 + 0];
  float qy = xyz[(b * n + f) * 3 + 1];
  float qz = xyz[(b * n + f) * 3 + 2];
  nxyz[t * 3 + 0] = qx; nxyz[t * 3 + 1] = qy; nxyz[t * 3 + 2] = qz;
  float bd[24]; int bi[24];
  for (int k = 0; k < 24; ++k) { bd[k] = 3e38f; bi[k] = 0; }
  for (int j = 0; j < n; ++j) {
    float dx = xyz[(b * n + j) * 3 + 0] - qx;
    float dy = xyz[(b * n + j) * 3 + 1] - qy;
    float dz = xyz[(b * n + j) * 3 + 2] - qz;
    float d = dx * dx + dy * dy + dz * dz;
    if (d < bd[23]) {
      int k = 23;
      while (k > 0 && bd[k - 1] > d) { bd[k] = bd[k - 1]; bi[k] = bi[k - 1]; --k; }
      bd[k] = d; bi[k] = j;
    }
  }
  for (int k = 0; k < 24; ++k) idx[(long)t * 24 + k] = bi[k];
}

// ---------- per-batch sum/sumsq of diff (for std, ddof=1) ----------
__global__ void stats_kernel(const u16* __restrict__ h, const int* __restrict__ idx,
                             const int* __restrict__ fi, int n, int nnew, int C,
                             float* __restrict__ stats) {
  int b = blockIdx.y;
  long total = (long)nnew * 24 * C;
  float s = 0.f, sq = 0.f;
  for (long e = (long)blockIdx.x * blockDim.x + threadIdx.x; e < total;
       e += (long)gridDim.x * blockDim.x) {
    int c = (int)(e % C);
    long r = e / C;
    int sp = (int)(r % 24);
    int a  = (int)(r / 24);
    int pi = idx[((long)b * nnew + a) * 24 + sp];
    int fp = fi[b * nnew + a];
    float d = bf2f(h[((long)(b * n + pi)) * C + c]) -
              bf2f(h[((long)(b * n + fp)) * C + c]);
    s += d; sq += d * d;
  }
  __shared__ float ss[256], sg[256];
  ss[threadIdx.x] = s; sg[threadIdx.x] = sq;
  __syncthreads();
  for (int st = 128; st > 0; st >>= 1) {
    if (threadIdx.x < st) {
      ss[threadIdx.x] += ss[threadIdx.x + st];
      sg[threadIdx.x] += sg[threadIdx.x + st];
    }
    __syncthreads();
  }
  if (threadIdx.x == 0) {
    atomicAdd(&stats[b * 2 + 0], ss[0]);
    atomicAdd(&stats[b * 2 + 1], sg[0]);
  }
}

// ---------- group+normalize+concat pack: y[group][32][2C] bf16, rows 24..31 zero ----------
__global__ void pack_kernel(const u16* __restrict__ h, const int* __restrict__ idx,
                            const int* __restrict__ fi, const float* __restrict__ alpha,
                            const float* __restrict__ beta, const float* __restrict__ stats,
                            int n, int nnew, int C, u16* __restrict__ y) {
  long t = (long)blockIdx.x * blockDim.x + threadIdx.x;
  int d2 = 2 * C;
  long total = (long)16 * nnew * 32 * d2;
  if (t >= total) return;
  int ch = (int)(t % d2);
  long r = t / d2;
  int sp = (int)(r % 32);
  long grp = r / 32;
  int b = (int)(grp / nnew);
  int a = (int)(grp % nnew);
  u16 out = 0;
  if (sp < 24) {
    int fp = fi[b * nnew + a];
    if (ch < C) {
      long cnt = (long)nnew * 24 * C;
      float s = stats[b * 2 + 0], sq = stats[b * 2 + 1];
      float var = (sq - s * s / (float)cnt) / (float)(cnt - 1);
      float stdv = sqrtf(fmaxf(var, 0.f));
      float inv = 1.f / (stdv + EPSV);
      int pi = idx[((long)b * nnew + a) * 24 + sp];
      float dv = bf2f(h[((long)(b * n + pi)) * C + ch]) -
                 bf2f(h[((long)(b * n + fp)) * C + ch]);
      out = f2bf_bits(alpha[ch] * dv * inv + beta[ch]);
    } else {
      out = h[((long)(b * n + fp)) * C + (ch - C)]; // mean channel: copy bits
    }
  }
  y[t] = out;
}

// ---------- WMMA bf16 GEMM + fused BN (+residual) + ReLU ----------
// Wave computes a 32(M)x32(N) block: 2 M-tiles x 2 N-tiles -> 4 WMMA / K-step,
// A fragments shared across N-tiles, B fragments shared across M-tiles
// (2 b128 loads per WMMA instead of 3).
// Wt: bf16 [M][Kd] row-major.  X/Res/Out: bf16 [G][Npad][{Kd|M}] (N-major).
template <bool RESID>
__global__ __launch_bounds__(256) void gemm_bn_kernel(
    const u16* __restrict__ Wt, const u16* __restrict__ X,
    const u16* __restrict__ Res, u16* __restrict__ Out,
    const float* __restrict__ bng, const float* __restrict__ bnb,
    const float* __restrict__ bnm, const float* __restrict__ bnv,
    int G, int M, int Kd, int Npad, int Nvalid) {
  int wave = threadIdx.x >> 5;
  int lane = threadIdx.x & 31;
  int mt2 = M >> 5, nt = Npad >> 5;
  long task = (long)blockIdx.x * 8 + wave;
  long tasks = (long)G * mt2 * nt;
  if (task >= tasks) return;
  int g = (int)(task / (mt2 * nt));
  int rem = (int)(task % (mt2 * nt));
  int mtile = rem / nt, nchunk = rem % nt;
  int m0 = mtile << 5, nb = nchunk << 5;
  int hl = lane >> 4;       // half-wave id
  int l15 = lane & 15;      // A: row, B/D: column

  v8f acc00 = {}, acc01 = {}, acc10 = {}, acc11 = {};
  const u16* wrow0 = Wt + (size_t)(m0 + l15) * Kd + 16 * hl;
  const u16* wrow1 = wrow0 + (size_t)16 * Kd;
  const u16* xrow0 = X + ((size_t)g * Npad + nb + l15) * Kd + 16 * hl;
  const u16* xrow1 = xrow0 + (size_t)16 * Kd;

  for (int kk = 0; kk < Kd; kk += 32) {
    us8 a0lo = *(const us8*)(wrow0 + kk);
    us8 a0hi = *(const us8*)(wrow0 + kk + 8);
    us8 a1lo = *(const us8*)(wrow1 + kk);
    us8 a1hi = *(const us8*)(wrow1 + kk + 8);
    v16bf a0 = __builtin_bit_cast(
        v16bf, __builtin_shufflevector(a0lo, a0hi, 0, 1, 2, 3, 4, 5, 6, 7, 8, 9,
                                       10, 11, 12, 13, 14, 15));
    v16bf a1 = __builtin_bit_cast(
        v16bf, __builtin_shufflevector(a1lo, a1hi, 0, 1, 2, 3, 4, 5, 6, 7, 8, 9,
                                       10, 11, 12, 13, 14, 15));
    v16bf b0 = __builtin_bit_cast(v16bf, *(const us16*)(xrow0 + kk));
    v16bf b1 = __builtin_bit_cast(v16bf, *(const us16*)(xrow1 + kk));
    acc00 = __builtin_amdgcn_wmma_f32_16x16x32_bf16(false, a0, false, b0,
                                                    (short)0, acc00, false, false);
    acc01 = __builtin_amdgcn_wmma_f32_16x16x32_bf16(false, a0, false, b1,
                                                    (short)0, acc01, false, false);
    acc10 = __builtin_amdgcn_wmma_f32_16x16x32_bf16(false, a1, false, b0,
                                                    (short)0, acc10, false, false);
    acc11 = __builtin_amdgcn_wmma_f32_16x16x32_bf16(false, a1, false, b1,
                                                    (short)0, acc11, false, false);
  }

  int mb0 = m0 + 8 * hl;    // 8 consecutive output channels per lane, tile 0
  int mb1 = mb0 + 16;       // tile 1
  float sc0[8], bi0[8], sc1[8], bi1[8];
  for (int v = 0; v < 8; ++v) {
    float s0 = bng[mb0 + v] * rsqrtf(bnv[mb0 + v] + EPSV);
    sc0[v] = s0;
    bi0[v] = bnb[mb0 + v] - bnm[mb0 + v] * s0;
    float s1 = bng[mb1 + v] * rsqrtf(bnv[mb1 + v] + EPSV);
    sc1[v] = s1;
    bi1[v] = bnb[mb1 + v] - bnm[mb1 + v] * s1;
  }
  int N0 = nb + l15, N1 = nb + 16 + l15;

  auto epi = [&](const v8f& acc, int mbase, const float* sc, const float* bi,
                 int N) {
    size_t o = ((size_t)g * Npad + N) * M + mbase;
    us8 rres = {};
    if constexpr (RESID) rres = *(const us8*)(Res + o);
    us8 outv = {};
    for (int v = 0; v < 8; ++v) {
      float val = acc[v] * sc[v] + bi[v];
      if constexpr (RESID) val += bf2f(rres[v]);
      val = fmaxf(val, 0.f);
      if (N >= Nvalid) val = 0.f;
      outv[v] = f2bf_bits(val);
    }
    *(us8*)(Out + o) = outv;
  };
  epi(acc00, mb0, sc0, bi0, N0);
  epi(acc01, mb0, sc0, bi0, N1);
  epi(acc10, mb1, sc1, bi1, N0);
  epi(acc11, mb1, sc1, bi1, N1);
}

// ---------- max over s (24 valid of 32) : z[G][32][C] -> h[G][C] ----------
__global__ void maxpool_s_kernel(const u16* __restrict__ z, long G, int C,
                                 u16* __restrict__ h) {
  long t = (long)blockIdx.x * blockDim.x + threadIdx.x;
  if (t >= G * C) return;
  int c = (int)(t % C);
  long g = t / C;
  float m = -3e38f;
  for (int sp = 0; sp < 24; ++sp) m = fmaxf(m, bf2f(z[(g * 32 + sp) * C + c]));
  h[g * C + c] = f2bf_bits(m);
}

// ---------- final max over points: h[b][64][1024] -> f[b][1024] ----------
__global__ void maxpool_n_kernel(const u16* __restrict__ h, float* __restrict__ f) {
  int t = blockIdx.x * blockDim.x + threadIdx.x;
  if (t >= 16 * 1024) return;
  int b = t >> 10, c = t & 1023;
  float m = -3e38f;
  for (int p = 0; p < 64; ++p)
    m = fmaxf(m, bf2f(h[((long)(b * 64 + p)) * 1024 + c]));
  f[t] = m;
}

// ---------- classifier head: one block per batch element ----------
__global__ void head_kernel(const float* __restrict__ f,
                            const float* __restrict__ W1, const float* __restrict__ b1,
                            const float* __restrict__ g1, const float* __restrict__ bb1,
                            const float* __restrict__ m1, const float* __restrict__ v1,
                            const float* __restrict__ W2, const float* __restrict__ b2,
                            const float* __restrict__ g2, const float* __restrict__ bb2,
                            const float* __restrict__ m2, const float* __restrict__ v2,
                            const float* __restrict__ W3, const float* __restrict__ b3,
                            float* __restrict__ out) {
  int b = blockIdx.x, tid = threadIdx.x;
  __shared__ float sf[1024];
  __shared__ float s1[512];
  __shared__ float s2[256];
  for (int c = tid; c < 1024; c += 512) sf[c] = f[b * 1024 + c];
  __syncthreads();
  {
    int o = tid;
    float acc = b1[o];
    for (int c = 0; c < 1024; ++c) acc += sf[c] * W1[(long)o * 1024 + c];
    float s = g1[o] * rsqrtf(v1[o] + EPSV);
    s1[o] = fmaxf(acc * s + (bb1[o] - m1[o] * s), 0.f);
  }
  __syncthreads();
  if (tid < 256) {
    int o = tid;
    float acc = b2[o];
    for (int c = 0; c < 512; ++c) acc += s1[c] * W2[(long)o * 512 + c];
    float s = g2[o] * rsqrtf(v2[o] + EPSV);
    s2[o] = fmaxf(acc * s + (bb2[o] - m2[o] * s), 0.f);
  }
  __syncthreads();
  if (tid < 40) {
    int o = tid;
    float acc = b3[o];
    for (int c = 0; c < 256; ++c) acc += s2[c] * W3[(long)o * 256 + c];
    out[b * 40 + o] = acc;
  }
}

// =====================================================================

static void launch_gemm(bool resid, const u16* Wt, const u16* X, const u16* Res,
                        u16* Out, const float* g, const float* bb, const float* m,
                        const float* v, int G, int M, int Kd, int Npad, int Nvalid,
                        hipStream_t stream) {
  long tasks = (long)G * (M / 32) * (Npad / 32);
  int blocks = (int)((tasks + 7) / 8);
  if (resid)
    gemm_bn_kernel<true><<<blocks, 256, 0, stream>>>(Wt, X, Res, Out, g, bb, m, v,
                                                     G, M, Kd, Npad, Nvalid);
  else
    gemm_bn_kernel<false><<<blocks, 256, 0, stream>>>(Wt, X, Res, Out, g, bb, m, v,
                                                      G, M, Kd, Npad, Nvalid);
}

extern "C" void kernel_launch(void* const* d_in, const int* in_sizes, int n_in,
                              void* d_out, int out_size, void* d_ws, size_t ws_size,
                              hipStream_t stream) {
  (void)in_sizes; (void)n_in; (void)out_size; (void)ws_size;
  const float* P[256];
  for (int i = 0; i < n_in && i < 256; ++i) P[i] = (const float*)d_in[i];

  // --- parameter map (jax pytree order: sorted dict keys, lists in order) ---
  const float* x = P[0];
  const float *clsW1 = P[1], *clsW2 = P[2], *clsW3 = P[3];
  const float *clsb1 = P[4], *clsb2 = P[5], *clsb3 = P[6];
  const float *c1b = P[7], *c1g = P[8], *c1m = P[9], *c1v = P[10];
  const float *c2b = P[11], *c2g = P[12], *c2m = P[13], *c2v = P[14];
  const float* embW = P[15];
  const float *eb = P[16], *eg = P[17], *em = P[18], *ev = P[19];

  const int Cp[4]   = {64, 128, 256, 512};
  const int Co[4]   = {128, 256, 512, 1024};
  const int Nin[4]  = {1024, 512, 256, 128};
  const int Nnew[4] = {512, 256, 128, 64};

  // per-stage parameter pointers
  const float *alpha[4], *beta[4], *tW[4];
  const float *tbn[4][4];                 // b,g,m,v
  const float *preW1[4][2], *preW2[4][2], *posW1[4][2], *posW2[4][2];
  const float *preBN1[4][2][4], *preBN2[4][2][4];
  const float *posBN1[4][2][4], *posBN2[4][2][4];
  for (int i = 0; i < 4; ++i) {
    int base = 20 + i * 47;
    alpha[i] = P[base + 0];
    beta[i]  = P[base + 1];
    for (int j = 0; j < 2; ++j) {
      int o = base + 2 + j * 10;          // pos blocks
      posW1[i][j] = P[o + 0];
      posW2[i][j] = P[o + 1];
      for (int q = 0; q < 4; ++q) posBN1[i][j][q] = P[o + 2 + q];
      for (int q = 0; q < 4; ++q) posBN2[i][j][q] = P[o + 6 + q];
    }
    for (int j = 0; j < 2; ++j) {
      int o = base + 22 + j * 10;         // pre blocks
      preW1[i][j] = P[o + 0];
      preW2[i][j] = P[o + 1];
      for (int q = 0; q < 4; ++q) preBN1[i][j][q] = P[o + 2 + q];
      for (int q = 0; q < 4; ++q) preBN2[i][j][q] = P[o + 6 + q];
    }
    tW[i] = P[base + 42];
    for (int q = 0; q < 4; ++q) tbn[i][q] = P[base + 43 + q];
  }

  // --- workspace carving (256B aligned) ---
  char* wbase = (char*)d_ws;
  size_t off = 0;
  auto carve = [&](size_t bytes) -> char* {
    char* p = wbase + off;
    off = (off + bytes + 255) & ~(size_t)255;
    return p;
  };
  float* xyzA = (float*)carve(16 * 1024 * 3 * sizeof(float));
  float* xyzB = (float*)carve(16 * 1024 * 3 * sizeof(float));
  int* fi   = (int*)carve(16 * 512 * sizeof(int));
  int* idxb = (int*)carve((size_t)16 * 512 * 24 * sizeof(int));
  float* stats = (float*)carve(16 * 2 * sizeof(float));
  float* fbuf  = (float*)carve(16 * 1024 * sizeof(float));
  u16* hbuf[4];
  for (int i = 0; i < 4; ++i) hbuf[i] = (u16*)carve((size_t)16 * 1024 * 64 * 2);
  u16* bufA = (u16*)carve((size_t)33554432 * 2);
  u16* bufB = (u16*)carve((size_t)33554432 * 2);
  u16* bufC = (u16*)carve((size_t)33554432 * 2);
  u16 *wT[4], *wPre1[4][2], *wPre2[4][2], *wPos1[4][2], *wPos2[4][2];
  for (int i = 0; i < 4; ++i) {
    size_t ne = (size_t)Co[i] * Co[i];
    wT[i] = (u16*)carve(ne * 2);
    for (int j = 0; j < 2; ++j) {
      wPre1[i][j] = (u16*)carve(ne * 2);
      wPre2[i][j] = (u16*)carve(ne * 2);
      wPos1[i][j] = (u16*)carve(ne * 2);
      wPos2[i][j] = (u16*)carve(ne * 2);
    }
  }

  // --- convert conv weights to bf16 ---
  auto cvt = [&](const float* src, u16* dst, long n) {
    convert_f32_bf16_kernel<<<(int)((n + 255) / 256), 256, 0, stream>>>(src, dst, n);
  };
  for (int i = 0; i < 4; ++i) {
    long ne = (long)Co[i] * Co[i];
    cvt(tW[i], wT[i], ne);
    for (int j = 0; j < 2; ++j) {
      cvt(preW1[i][j], wPre1[i][j], ne);
      cvt(preW2[i][j], wPre2[i][j], ne);
      cvt(posW1[i][j], wPos1[i][j], ne);
      cvt(posW2[i][j], wPos2[i][j], ne);
    }
  }

  // --- embed + xyz transpose ---
  embed_kernel<<<(16 * 1024) / 256, 256, 0, stream>>>(x, embW, eb, eg, em, ev,
                                                      hbuf[0], xyzA);
  u16* h = hbuf[0];
  float* xyzCur = xyzA;
  float* xyzNext = xyzB;
  int hcur = 0;

  for (int i = 0; i < 4; ++i) {
    int n = Nin[i], nn = Nnew[i], C = Cp[i], O = Co[i];
    // FPS + KNN (+ gather new_xyz)
    fps_kernel<<<16, 256, 0, stream>>>(xyzCur, n, nn, fi);
    knn_kernel<<<(16 * nn + 127) / 128, 128, 0, stream>>>(xyzCur, fi, n, nn,
                                                          xyzNext, idxb);
    // normalization stats
    hipMemsetAsync(stats, 0, 16 * 2 * sizeof(float), stream);
    stats_kernel<<<dim3(256, 16), 256, 0, stream>>>(h, idxb, fi, n, nn, C, stats);
    // pack groups -> bufA : [G][32][2C] bf16
    {
      long total = (long)16 * nn * 32 * (2 * C);
      pack_kernel<<<(int)((total + 255) / 256), 256, 0, stream>>>(
          h, idxb, fi, alpha[i], beta[i], stats, n, nn, C, bufA);
    }
    long G = (long)16 * nn;
    // transfer: A -> B
    launch_gemm(false, wT[i], bufA, nullptr, bufB, tbn[i][1], tbn[i][0], tbn[i][2],
                tbn[i][3], (int)G, O, O, 32, 24, stream);
    // pre block 0: B -> C ; C (+B) -> A
    launch_gemm(false, wPre1[i][0], bufB, nullptr, bufC, preBN1[i][0][1],
                preBN1[i][0][0], preBN1[i][0][2], preBN1[i][0][3], (int)G, O, O,
                32, 24, stream);
    launch_gemm(true, wPre2[i][0], bufC, bufB, bufA, preBN2[i][0][1],
                preBN2[i][0][0], preBN2[i][0][2], preBN2[i][0][3], (int)G, O, O,
                32, 24, stream);
    // pre block 1: A -> C ; C (+A) -> B
    launch_gemm(false, wPre1[i][1], bufA, nullptr, bufC, preBN1[i][1][1],
                preBN1[i][1][0], preBN1[i][1][2], preBN1[i][1][3], (int)G, O, O,
                32, 24, stream);
    launch_gemm(true, wPre2[i][1], bufC, bufA, bufB, preBN2[i][1][1],
                preBN2[i][1][0], preBN2[i][1][2], preBN2[i][1][3], (int)G, O, O,
                32, 24, stream);
    // maxpool over s=24 -> h1 [b][nn][O]
    u16* h1 = hbuf[(hcur + 1) & 3];
    u16* h2 = hbuf[(hcur + 2) & 3];
    u16* h3 = hbuf[(hcur + 3) & 3];
    {
      long total = G * O;
      maxpool_s_kernel<<<(int)((total + 255) / 256), 256, 0, stream>>>(bufB, G, O, h1);
    }
    // pos block 0: h1 -> h2 ; h2 (+h1) -> h3
    launch_gemm(false, wPos1[i][0], h1, nullptr, h2, posBN1[i][0][1],
                posBN1[i][0][0], posBN1[i][0][2], posBN1[i][0][3], 16, O, O, nn,
                nn, stream);
    launch_gemm(true, wPos2[i][0], h2, h1, h3, posBN2[i][0][1], posBN2[i][0][0],
                posBN2[i][0][2], posBN2[i][0][3], 16, O, O, nn, nn, stream);
    // pos block 1: h3 -> h2 ; h2 (+h3) -> h1
    launch_gemm(false, wPos1[i][1], h3, nullptr, h2, posBN1[i][1][1],
                posBN1[i][1][0], posBN1[i][1][2], posBN1[i][1][3], 16, O, O, nn,
                nn, stream);
    launch_gemm(true, wPos2[i][1], h2, h3, h1, posBN2[i][1][1], posBN2[i][1][0],
                posBN2[i][1][2], posBN2[i][1][3], 16, O, O, nn, nn, stream);
    h = h1;
    hcur = (hcur + 1) & 3;
    // swap xyz ping-pong
    float* tmp = xyzCur; xyzCur = xyzNext; xyzNext = tmp;
  }

  // global max pool + classifier head
  maxpool_n_kernel<<<(16 * 1024) / 256, 256, 0, stream>>>(h, fbuf);
  head_kernel<<<16, 512, 0, stream>>>(fbuf, clsW1, clsb1, c1g, c1b, c1m, c1v,
                                      clsW2, clsb2, c2g, c2b, c2m, c2v, clsW3,
                                      clsb3, (float*)d_out);
}
